// ImprovedAttention_88158498718436
// MI455X (gfx1250) — matrix operand order repaired and checked by reading, not compiled, not run on hardware
//
#include <hip/hip_runtime.h>
#include <hip/hip_bf16.h>

typedef __bf16 bf16;
typedef __attribute__((ext_vector_type(16))) __bf16 v16bf;
typedef __attribute__((ext_vector_type(8)))  __bf16 v8bf;
typedef __attribute__((ext_vector_type(8)))  float  v8f;

#define WMMA_BF16(a, b, c) \
  __builtin_amdgcn_wmma_f32_16x16x32_bf16(false, (a), false, (b), (short)0, (c), false, false)

// ---------------------------------------------------------------------------
// Fragment loaders (wave32), contiguous k order shared by A and B
// A 16x32 bf16 (row-major LDS, ld elems): lane row = lane&15; kh = lane>>4.
//   elems 0..15 -> K = kh*16 + e      (two 16B chunks)
// B 32x16 bf16 from COLUMN-MAJOR storage (col stride ldk): lane col = lane&15,
//   elems e -> K = kh*16 + e  == one contiguous 32B vector load per lane.
// C/D 16x16 f32: col = lane&15, row = (lane>>4)*8 + vgpr_index
// ---------------------------------------------------------------------------
__device__ __forceinline__ v16bf load_a_lds(const bf16* base, int ld) {
  const int lane = threadIdx.x & 31;
  const int row  = lane & 15;
  const int kh   = (lane >> 4) & 1;
  v8bf lo = *(const v8bf*)(base + row * ld + kh * 16);
  v8bf hi = *(const v8bf*)(base + row * ld + kh * 16 + 8);
  v16bf a;
#pragma unroll
  for (int e = 0; e < 8; ++e) { a[e] = lo[e]; a[8 + e] = hi[e]; }
  return a;
}

// base = &Bcm[frag_col0 * ldk + k_offset]; ldk = column stride (elems, mult of 16)
__device__ __forceinline__ v16bf load_b_cm(const bf16* base, int ldk) {
  const int lane = threadIdx.x & 31;
  const int col  = lane & 15;
  const int kh   = (lane >> 4) & 1;
  return *(const v16bf*)(base + col * ldk + kh * 16);
}

// ---------------------------------------------------------------------------
// fp32 -> bf16 conversion
// ---------------------------------------------------------------------------
__global__ void __launch_bounds__(256) cvt_f32_bf16(const float* __restrict__ src,
                                                    bf16* __restrict__ dst, int n) {
  int i = blockIdx.x * 256 + threadIdx.x;
  if (i < n) dst[i] = (bf16)src[i];
}

// ---------------------------------------------------------------------------
// GEMM: out[M][N] = A[M][K] @ W[N][K]^T   (A,W bf16; acc f32)
// block = 256 threads (8 waves), tile 128(M) x 64(N), k-step 32
// Double-buffered LDS (ping-pong): stage tile k+1 while WMMAs consume tile k;
// one barrier per k-step. Prefetch tile k+2 toward L2 (global_prefetch_b8).
// B fragments preloaded into registers so LDS loads batch ahead of WMMAs.
// ---------------------------------------------------------------------------
template <bool OUT_F32>
__global__ void __launch_bounds__(256)
gemm_wmma(const bf16* __restrict__ A, const bf16* __restrict__ W,
          const float* __restrict__ bias, float* __restrict__ outf,
          bf16* __restrict__ outb, int M, int N, int K) {
  __shared__ __attribute__((aligned(32))) bf16 As[2][128 * 40]; // row-major, ld 40
  __shared__ __attribute__((aligned(32))) bf16 Bs[2][64 * 32];  // col-major, ld 32
  const int tid  = threadIdx.x;
  const int wave = tid >> 5;
  const int m0 = blockIdx.y * 128;
  const int n0 = blockIdx.x * 64;

  auto stage = [&](int buf, int k0) {
#pragma unroll
    for (int i = 0; i < 16; ++i) {               // A tile: 128x32 row-major
      int idx = tid + i * 256;
      int r = idx >> 5, c = idx & 31;
      As[buf][r * 40 + c] = A[(size_t)(m0 + r) * K + k0 + c];
    }
#pragma unroll
    for (int i = 0; i < 8; ++i) {                // W tile -> Bs[n][kk] col-major
      int idx = tid + i * 256;
      int n = idx >> 5, kk = idx & 31;
      Bs[buf][n * 32 + kk] = W[(size_t)(n0 + n) * K + k0 + kk];
    }
  };

  v8f acc[4] = {};
  const int KT = K >> 5;

  stage(0, 0);
  for (int kt = 0; kt < KT; ++kt) {
    const int cur = kt & 1;
    __syncthreads();
    if (kt + 1 < KT) stage(cur ^ 1, (kt + 1) * 32);
    if (kt + 2 < KT) {                           // prefetch tile k+2 toward L2
      if (tid < 128)
        __builtin_prefetch(A + (size_t)(m0 + tid) * K + (kt + 2) * 32, 0, 1);
      else if (tid < 192)
        __builtin_prefetch(W + (size_t)(n0 + tid - 128) * K + (kt + 2) * 32, 0, 1);
    }
    v16bf a = load_a_lds(&As[cur][(wave * 16) * 40], 40);
    v16bf bfr[4];
#pragma unroll
    for (int nf = 0; nf < 4; ++nf)
      bfr[nf] = load_b_cm(&Bs[cur][(nf * 16) * 32], 32);
#pragma unroll
    for (int nf = 0; nf < 4; ++nf)
      acc[nf] = WMMA_BF16(a, bfr[nf], acc[nf]);
  }

  const int lane  = tid & 31;
  const int rbase = (lane >> 4) * 8;
  const int cn    = lane & 15;
#pragma unroll
  for (int nf = 0; nf < 4; ++nf) {
#pragma unroll
    for (int r = 0; r < 8; ++r) {
      int row = m0 + wave * 16 + rbase + r;
      int col = n0 + nf * 16 + cn;
      float v = acc[nf][r];
      if constexpr (OUT_F32) outf[(size_t)row * N + col] = v + bias[col];
      else                   outb[(size_t)row * N + col] = (bf16)v;
    }
  }
}

// ---------------------------------------------------------------------------
// RoPE + reorder: qkv (B,T,3,H,D) bf16 -> qr/kr (B,H,T,D) bf16
// ---------------------------------------------------------------------------
__global__ void __launch_bounds__(256)
rope_reorder(const bf16* __restrict__ qkvb, bf16* __restrict__ qr,
             bf16* __restrict__ kr) {
  constexpr int T = 2048;
  int idx = blockIdx.x * 256 + threadIdx.x;    // 0 .. 131071
  int t  = idx & (T - 1);
  int bh = (idx >> 11) & 31;
  int z  = idx >> 16;                          // 0 = q, 1 = k
  int b = bh >> 4, h = bh & 15;
  const bf16* src = qkvb + ((size_t)(b * T + t)) * 3072 + z * 1024 + h * 64;
  bf16* dst = (z ? kr : qr) + ((size_t)bh * T + t) * 64;
#pragma unroll
  for (int i = 0; i < 32; ++i) {
    float x1 = (float)src[2 * i];
    float x2 = (float)src[2 * i + 1];
    float freq = (float)t * __powf(10000.0f, -(float)(2 * i) / 64.0f);
    float sn, cs;
    __sincosf(freq, &sn, &cs);
    dst[i]      = (bf16)(x1 * cs - x2 * sn);
    dst[32 + i] = (bf16)(x1 * sn + x2 * cs);
  }
}

// ---------------------------------------------------------------------------
// Flash attention: grid (T/64, B*H), 128 threads (4 waves x 16 q-rows),
// KV tile = 64. All 8 K B-fragments preloaded into registers (16 clause'd
// global_load_b128) before the 8 S-WMMAs so loads drain with partial
// s_wait_loadcnt. V staged column-major into double-buffered LDS. P bounced
// via wave-private LDS (C-layout -> A-layout). Softmax via lane shuffles.
// ---------------------------------------------------------------------------
__global__ void __launch_bounds__(128)
attn_wmma(const bf16* __restrict__ qr, const bf16* __restrict__ kr,
          const bf16* __restrict__ qkvb, bf16* __restrict__ attnb) {
  constexpr int T = 2048, D = 64;
  __shared__ __attribute__((aligned(32))) bf16 Vs[2][64 * 64];   // col-major [d][kk]
  __shared__ __attribute__((aligned(32))) bf16 Ps[4 * 16 * 64];  // per-wave P 16x64
  const int tid  = threadIdx.x;
  const int wave = tid >> 5;
  const int lane = tid & 31;
  const int bh = blockIdx.y;
  const int b  = bh >> 4, h = bh & 15;
  const int q0 = blockIdx.x * 64 + wave * 16;

  const int col = lane & 15;
  const int kh  = (lane >> 4) & 1;
  const int rbase = kh * 8;

  // Q fragments for the two D k-steps, loaded once (global, 16B chunks)
  v16bf aq[2];
  {
    const bf16* qrow = qr + ((size_t)bh * T + q0 + col) * D;
#pragma unroll
    for (int ks = 0; ks < 2; ++ks) {
      v8bf lo = *(const v8bf*)(qrow + ks * 32 + kh * 16);
      v8bf hi = *(const v8bf*)(qrow + ks * 32 + kh * 16 + 8);
#pragma unroll
      for (int e = 0; e < 8; ++e) { aq[ks][e] = lo[e]; aq[ks][8 + e] = hi[e]; }
    }
  }

  auto stageV = [&](int buf, int kv0) {
#pragma unroll
    for (int i = 0; i < 32; ++i) {
      int idx = tid + i * 128;                 // 0..4095
      int kk = idx >> 6, d = idx & 63;
      Vs[buf][d * 64 + kk] =
          qkvb[((size_t)(b * T + kv0 + kk)) * 3072 + 2048 + h * 64 + d];
    }
  };

  float mrun[8], lrun[8];
#pragma unroll
  for (int r = 0; r < 8; ++r) { mrun[r] = -1e30f; lrun[r] = 0.0f; }
  v8f o[4] = {};

  const float scale = 0.125f;                  // D^-0.5, D=64
  bf16* myP = &Ps[wave * 16 * 64];
  constexpr int NT = T / 64;

  stageV(0, 0);
  for (int it = 0; it < NT; ++it) {
    const int kv0 = it * 64;
    const int cur = it & 1;
    __syncthreads();
    if (it + 1 < NT) {
      stageV(cur ^ 1, kv0 + 64);               // overlap with WMMA on cur
      if (tid < 64)                            // prefetch next K rows
        __builtin_prefetch(kr + ((size_t)bh * T + kv0 + 64 + tid) * D, 0, 1);
    }

    // preload all 8 K B-fragments (distinct registers -> batched loads)
    v16bf bk[8];
    {
      const bf16* kbase = kr + ((size_t)bh * T + kv0 + col) * D + kh * 16;
#pragma unroll
      for (int nf = 0; nf < 4; ++nf)
#pragma unroll
        for (int ks = 0; ks < 2; ++ks)
          bk[nf * 2 + ks] = *(const v16bf*)(kbase + nf * (16 * D) + ks * 32);
    }

    // S = Q K^T  (four 16-col N-fragments, two k-steps over D)
    v8f s[4] = {};
#pragma unroll
    for (int nf = 0; nf < 4; ++nf)
#pragma unroll
      for (int ks = 0; ks < 2; ++ks)
        s[nf] = WMMA_BF16(aq[ks], bk[nf * 2 + ks], s[nf]);

    // online softmax (row stats shared by the 16 lanes of each half-wave)
    float corr[8];
#pragma unroll
    for (int r = 0; r < 8; ++r) {
      float v0 = s[0][r] * scale, v1 = s[1][r] * scale;
      float v2 = s[2][r] * scale, v3 = s[3][r] * scale;
      float mx = fmaxf(fmaxf(v0, v1), fmaxf(v2, v3));
#pragma unroll
      for (int msk = 1; msk < 16; msk <<= 1)
        mx = fmaxf(mx, __shfl_xor(mx, msk, 32));
      float mnew = fmaxf(mrun[r], mx);
      corr[r] = __expf(mrun[r] - mnew);
      float p0 = __expf(v0 - mnew), p1 = __expf(v1 - mnew);
      float p2 = __expf(v2 - mnew), p3 = __expf(v3 - mnew);
      s[0][r] = p0; s[1][r] = p1; s[2][r] = p2; s[3][r] = p3;
      float ps = (p0 + p1) + (p2 + p3);
#pragma unroll
      for (int msk = 1; msk < 16; msk <<= 1)
        ps += __shfl_xor(ps, msk, 32);
      lrun[r] = lrun[r] * corr[r] + ps;
      mrun[r] = mnew;
    }

    // transpose P (C-layout) -> wave-private LDS row-major 16x64
#pragma unroll
    for (int nf = 0; nf < 4; ++nf)
#pragma unroll
      for (int r = 0; r < 8; ++r)
        myP[(rbase + r) * 64 + nf * 16 + col] = (bf16)s[nf][r];

#pragma unroll
    for (int nf = 0; nf < 4; ++nf)
#pragma unroll
      for (int r = 0; r < 8; ++r) o[nf][r] *= corr[r];

    // O += P @ V  (two k-steps over the 64 kv positions)
#pragma unroll
    for (int ks = 0; ks < 2; ++ks) {
      v16bf ap = load_a_lds(myP + ks * 32, 64);
      v16bf bv[4];
#pragma unroll
      for (int nf = 0; nf < 4; ++nf)
        bv[nf] = load_b_cm(&Vs[cur][(nf * 16) * 64] + ks * 32, 64);
#pragma unroll
      for (int nf = 0; nf < 4; ++nf)
        o[nf] = WMMA_BF16(ap, bv[nf], o[nf]);
    }
  }

  // finalize and write attn output (B,T,H*D) bf16
#pragma unroll
  for (int nf = 0; nf < 4; ++nf)
#pragma unroll
    for (int r = 0; r < 8; ++r) {
      int trow = q0 + rbase + r;
      int d    = nf * 16 + col;
      attnb[((size_t)(b * T + trow)) * 1024 + h * 64 + d] =
          (bf16)(o[nf][r] / lrun[r]);
    }
}

// ---------------------------------------------------------------------------
// Host-side orchestration
// ---------------------------------------------------------------------------
extern "C" void kernel_launch(void* const* d_in, const int* in_sizes, int n_in,
                              void* d_out, int out_size, void* d_ws, size_t ws_size,
                              hipStream_t stream) {
  const float* x      = (const float*)d_in[0];
  const float* w_qkv  = (const float*)d_in[1];
  const float* w_proj = (const float*)d_in[2];
  const float* b_proj = (const float*)d_in[3];
  float* out = (float*)d_out;

  const size_t NX  = 4096ull * 1024;   // x elements (B*T, C)
  const size_t NWQ = 3072ull * 1024;   // w_qkv
  const size_t NWP = 1024ull * 1024;   // w_proj
  const size_t NQKV = 4096ull * 3072;  // qkv activations
  const size_t NHD = 2ull * 16 * 2048 * 64; // per q/k reordered tensor

  bf16* xb     = (bf16*)d_ws;
  bf16* wqkvb  = xb + NX;
  bf16* wprojb = wqkvb + NWQ;
  bf16* qkvb   = wprojb + NWP;
  bf16* qrr    = qkvb + NQKV;
  bf16* krr    = qrr + NHD;
  bf16* attnb  = krr + NHD;

  cvt_f32_bf16<<<(int)((NX + 255) / 256), 256, 0, stream>>>(x, xb, (int)NX);
  cvt_f32_bf16<<<(int)((NWQ + 255) / 256), 256, 0, stream>>>(w_qkv, wqkvb, (int)NWQ);
  cvt_f32_bf16<<<(int)((NWP + 255) / 256), 256, 0, stream>>>(w_proj, wprojb, (int)NWP);

  // qkv = x @ w_qkv^T  : M=4096, N=3072, K=1024
  gemm_wmma<false><<<dim3(3072 / 64, 4096 / 128), 256, 0, stream>>>(
      xb, wqkvb, nullptr, nullptr, qkvb, 4096, 3072, 1024);

  rope_reorder<<<131072 / 256, 256, 0, stream>>>(qkvb, qrr, krr);

  attn_wmma<<<dim3(2048 / 64, 32), 128, 0, stream>>>(qrr, krr, qkvb, attnb);

  // out = attn @ w_proj^T + b  : M=4096, N=1024, K=1024
  gemm_wmma<true><<<dim3(1024 / 64, 4096 / 128), 256, 0, stream>>>(
      attnb, wprojb, b_proj, out, nullptr, 4096, 1024, 1024);
}
